// QueryGuidedMoE_40312563040758
// MI455X (gfx1250) — matrix operand order repaired and checked by reading, not compiled, not run on hardware
//
#include <hip/hip_runtime.h>
#include <stddef.h>

// ---------------------------------------------------------------------------
// Types for WMMA / TDM
// ---------------------------------------------------------------------------
typedef __attribute__((ext_vector_type(16))) __bf16 v16bf;
typedef __attribute__((ext_vector_type(8)))  float  v8f;
typedef __attribute__((ext_vector_type(4)))  unsigned int u32x4;
typedef __attribute__((ext_vector_type(8)))  int    i32x8;
typedef __attribute__((ext_vector_type(4)))  int    i32x4;

union FragBF {
    v16bf v;
    unsigned short h[16];
    unsigned int   u[8];
};
union FragF {
    v8f   v;
    float f[8];
};

// bf16 <-> f32 helpers (round-to-nearest-even, no header dependence)
__device__ __forceinline__ unsigned short f2bf(float f) {
    unsigned u = __builtin_bit_cast(unsigned, f);
    unsigned r = (u + 0x7FFFu + ((u >> 16) & 1u)) >> 16;
    return (unsigned short)r;
}
__device__ __forceinline__ float bf2f(unsigned short h) {
    unsigned u = ((unsigned)h) << 16;
    return __builtin_bit_cast(float, u);
}

#define B_TOK   8192
#define HDIM    1024
#define EEXP    8
#define P2DIM   128
#define NREG    5
#define NSH     2

#define USE_TDM (__has_builtin(__builtin_amdgcn_tensor_load_to_lds))

// ---------------------------------------------------------------------------
// f32 -> bf16 convert kernel
// ---------------------------------------------------------------------------
__global__ __launch_bounds__(256) void cvt_bf16_kernel(const float* __restrict__ in,
                                                       unsigned short* __restrict__ out,
                                                       size_t n) {
    size_t i = (size_t)blockIdx.x * 256 + threadIdx.x;
    if (i < n) out[i] = f2bf(in[i]);
}

// ---------------------------------------------------------------------------
// LayerNorm -> bf16 (strided output so it can land inside the concat buffer)
// ---------------------------------------------------------------------------
__global__ __launch_bounds__(256) void ln_kernel(const float* __restrict__ x,
                                                 const float* __restrict__ g,
                                                 const float* __restrict__ bb,
                                                 unsigned short* __restrict__ out,
                                                 int ldo) {
    const int row = blockIdx.x;
    const int tid = threadIdx.x;
    __shared__ float s0[256], s1[256];
    float p0 = 0.f, p1 = 0.f;
    for (int h = tid; h < HDIM; h += 256) {
        float v = x[(size_t)row * HDIM + h];
        p0 += v; p1 += v * v;
    }
    s0[tid] = p0; s1[tid] = p1;
    __syncthreads();
    for (int off = 128; off > 0; off >>= 1) {
        if (tid < off) { s0[tid] += s0[tid + off]; s1[tid] += s1[tid + off]; }
        __syncthreads();
    }
    const float mu  = s0[0] * (1.0f / HDIM);
    const float var = s1[0] * (1.0f / HDIM) - mu * mu;
    const float rs  = rsqrtf(var + 1e-5f);
    for (int h = tid; h < HDIM; h += 256) {
        float v = (x[(size_t)row * HDIM + h] - mu) * rs * g[h] + bb[h];
        out[(size_t)row * ldo + h] = f2bf(v);
    }
}

// ---------------------------------------------------------------------------
// Tiled bf16 WMMA GEMM:  C[M,N] = epilogue(A[M,K] @ W[K,N] + bias)
//   A: bf16 row-major (lda), W: bf16 row-major (ldw == logical N stride)
//   block tile 128x64, 8 waves of 32x32 (2x2 WMMA 16x16x32), K step 32
//   A tile staged via Tensor Data Mover (TENSOR_LOAD_TO_LDS, wave 0 issues);
//   hardware applies the [128][40] LDS row padding (pad_interval=16 DWORDs,
//   pad_amount=4 DWORDs). W tile staged transposed for contiguous frag loads.
//   act: 0=none 1=relu 2=sigmoid
//   gates!=null: val *= gates[m*8+gidx]   (per-token expert gate)
//   accum!=0 (f32 out only): C += val  else C = val
// ---------------------------------------------------------------------------
template <bool OUT_BF16>
__global__ __launch_bounds__(256) void gemm_bf16_kernel(
    const unsigned short* __restrict__ A, int lda,
    const unsigned short* __restrict__ W, int ldw,
    const float* __restrict__ bias,
    void* __restrict__ C, int ldc,
    int K, int act,
    const float* __restrict__ gates, int gidx, int accum) {

    __shared__ __align__(16) unsigned short Asl[128][40];
    __shared__ __align__(16) unsigned short Bsl[64][40];

    const int tid   = threadIdx.x;
    const int lane  = tid & 31;
    const int wid   = tid >> 5;
    const int waveM = (wid >> 1) * 32;
    const int waveN = (wid & 1) * 32;
    const int blockM = blockIdx.y * 128;
    const int blockN = blockIdx.x * 64;
    const int hf = lane >> 4;     // which 16-lane half
    const int nl = lane & 15;

#if USE_TDM
    // ---- D# descriptor (uniform across the block): 2-D tile 32 x 128 ----
    const unsigned lds_base = (unsigned)(size_t)&Asl[0][0];
    // group1 word0: data_size=2B (code 1), pad_enable, pad_interval=16 DWORDs
    // (code 3), pad_amount=4 DWORDs (code 3)
    const unsigned g1w0 = (1u << 16) | (1u << 20) | (3u << 22) | (3u << 25);
    const i32x8 g1 = { (int)g1w0,
                       (int)(32u << 16),    // tensor_dim0[15:0] = 32
                       (int)(128u << 16),   // tensor_dim1[15:0] = 128
                       (int)(32u << 16),    // tile_dim0 = 32
                       (int)128u,           // tile_dim1 = 128
                       (int)(unsigned)lda,  // tensor_dim0_stride (elements)
                       0, 0 };
    const i32x4 z4 = { 0, 0, 0, 0 };
#if defined(__clang_major__) && (__clang_major__ >= 23)
    const i32x8 z8 = { 0, 0, 0, 0, 0, 0, 0, 0 };
#endif
#endif

    FragF c[2][2];
#pragma unroll
    for (int i = 0; i < 2; i++)
#pragma unroll
        for (int j = 0; j < 2; j++)
#pragma unroll
            for (int r = 0; r < 8; r++) c[i][j].f[r] = 0.f;

    for (int k0 = 0; k0 < K; k0 += 32) {
        __syncthreads();
#if USE_TDM
        // ---- stage A tile [128 x 32] via Tensor Data Mover (wave 0) ----
        if (wid == 0) {
            unsigned long long ga =
                (unsigned long long)(size_t)(A + (size_t)blockM * lda + k0);
            const u32x4 g0 = { 1u,                                  // count=1
                               lds_base,                            // lds_addr
                               (unsigned)ga,                        // gaddr lo
                               (unsigned)(ga >> 32) | 0x80000000u } // hi | type=2
                             ;
#if defined(__clang_major__) && (__clang_major__ >= 23)
            __builtin_amdgcn_tensor_load_to_lds(g0, g1, z4, z4, z8, 0);
#else
            __builtin_amdgcn_tensor_load_to_lds(g0, g1, z4, z4, 0);
#endif
        }
#else
        // ---- fallback: per-thread A tile staging (128b chunks) ----
#pragma unroll
        for (int it = 0; it < 2; ++it) {
            int slot = tid + it * 256;
            int r  = slot >> 2;
            int cc = slot & 3;
            const uint4* src = reinterpret_cast<const uint4*>(
                A + (size_t)(blockM + r) * lda + k0 + cc * 8);
            *reinterpret_cast<uint4*>(&Asl[r][cc * 8]) = *src;
        }
#endif
        // ---- stage W tile [32 x 64] transposed into Bsl[n][k] ----
        {
            int kk = tid >> 3;
            int cc = tid & 7;
            const uint4* src = reinterpret_cast<const uint4*>(
                W + (size_t)(k0 + kk) * ldw + blockN + cc * 8);
            uint4 wv = *src;
            const unsigned short* hp = reinterpret_cast<const unsigned short*>(&wv);
#pragma unroll
            for (int j = 0; j < 8; j++) Bsl[cc * 8 + j][kk] = hp[j];
        }
        // prefetch next A tile (global_prefetch_b8)
        if (k0 + 32 < K)
            __builtin_prefetch(A + (size_t)(blockM + (tid >> 1)) * lda + k0 + 32, 0, 3);
#if USE_TDM
        if (wid == 0) __builtin_amdgcn_s_wait_tensorcnt(0);
#endif
        __syncthreads();

        // ---- build fragments per ISA §7.12.2 bf16 layouts ----
        FragBF a[2], b[2];
#pragma unroll
        for (int mi = 0; mi < 2; mi++) {
            int row = waveM + mi * 16 + nl;
#pragma unroll
            for (int v = 0; v < 8; ++v) {
                int kk = (v < 4 ? 2 * v : 16 + 2 * (v - 4)) + 8 * hf;
                a[mi].u[v] = *reinterpret_cast<const unsigned*>(&Asl[row][kk]);
            }
        }
#pragma unroll
        for (int ni = 0; ni < 2; ni++) {
            int col = waveN + ni * 16 + nl;
#pragma unroll
            for (int v = 0; v < 8; ++v) {
                int kk = (v < 4 ? 2 * v : 16 + 2 * (v - 4)) + 8 * hf;
                b[ni].u[v] = *reinterpret_cast<const unsigned*>(&Bsl[col][kk]);
            }
        }
#pragma unroll
        for (int mi = 0; mi < 2; mi++)
#pragma unroll
            for (int ni = 0; ni < 2; ni++)
                c[mi][ni].v = __builtin_amdgcn_wmma_f32_16x16x32_bf16(
                    false, a[mi].v, false, b[ni].v, (short)0, c[mi][ni].v, false, false);
    }

    // ---- epilogue ----
    float*          Cf = reinterpret_cast<float*>(C);
    unsigned short* Cb = reinterpret_cast<unsigned short*>(C);
#pragma unroll
    for (int mi = 0; mi < 2; mi++)
#pragma unroll
        for (int ni = 0; ni < 2; ni++)
#pragma unroll
            for (int r = 0; r < 8; r++) {
                int m = blockM + waveM + mi * 16 + hf * 8 + r;
                int n = blockN + waveN + ni * 16 + nl;
                float val = c[mi][ni].f[r];
                if (bias) val += bias[n];
                if (act == 1)      val = fmaxf(val, 0.f);
                else if (act == 2) val = 1.f / (1.f + expf(-val));
                if (gates) val *= gates[(size_t)m * EEXP + gidx];
                size_t idx = (size_t)m * ldc + n;
                if (OUT_BF16) {
                    Cb[idx] = f2bf(val);
                } else {
                    if (accum) Cf[idx] += val;
                    else       Cf[idx] = val;
                }
            }
}

// ---------------------------------------------------------------------------
// Router: per-token tanh(fused@wg1)@wg2 -> softmax(E) -> top-2 -> dense gates
// one wave (32 threads) per token
// ---------------------------------------------------------------------------
__global__ __launch_bounds__(32) void router_kernel(
    const unsigned short* __restrict__ fused,   // [B,H] bf16
    const float* __restrict__ wg1,              // [H,32]
    const float* __restrict__ wg2,              // [32,8]
    float* __restrict__ gates) {                // [B,8]
    const int b = blockIdx.x;
    const int j = threadIdx.x;
    __shared__ float ts[32];
    __shared__ float lg[EEXP];

    float acc = 0.f;
    const size_t base = (size_t)b * HDIM;
    for (int h = 0; h < HDIM; ++h)
        acc += bf2f(fused[base + h]) * wg1[(size_t)h * 32 + j];
    ts[j] = tanhf(acc);
    __syncthreads();
    if (j < EEXP) {
        float l = 0.f;
        for (int jj = 0; jj < 32; ++jj) l += ts[jj] * wg2[jj * EEXP + j];
        lg[j] = l;
    }
    __syncthreads();
    if (j == 0) {
        float mx = lg[0];
        for (int e = 1; e < EEXP; ++e) mx = fmaxf(mx, lg[e]);
        float w[EEXP]; float s = 0.f;
        for (int e = 0; e < EEXP; ++e) { w[e] = expf(lg[e] - mx); s += w[e]; }
        float inv = 1.f / s;
        for (int e = 0; e < EEXP; ++e) w[e] *= inv;
        int i0 = 0;
        for (int e = 1; e < EEXP; ++e) if (w[e] > w[i0]) i0 = e;
        int i1 = (i0 == 0) ? 1 : 0;
        for (int e = 0; e < EEXP; ++e) if (e != i0 && w[e] > w[i1]) i1 = e;
        float norm = 1.f / (w[i0] + w[i1] + 1e-6f);
        for (int e = 0; e < EEXP; ++e) gates[(size_t)b * EEXP + e] = 0.f;
        gates[(size_t)b * EEXP + i0] = w[i0] * norm;
        gates[(size_t)b * EEXP + i1] = w[i1] * norm;
    }
}

// ---------------------------------------------------------------------------
// Zero-init routed slice of combined[:, 0:128] (ldc = 384)
// ---------------------------------------------------------------------------
__global__ __launch_bounds__(256) void zero_routed_kernel(float* __restrict__ combined) {
    int idx = blockIdx.x * 256 + threadIdx.x;
    int b = idx >> 7;
    int n = idx & 127;
    combined[(size_t)b * 384 + n] = 0.f;
}

// ---------------------------------------------------------------------------
// Constant + copy experts:
//   cw = softmax2(x @ ce_wg); combined[:, :128] += g5*(cw0*tmp1 + cw1*const) + g6*tmp2
// one block of 128 threads per token
// ---------------------------------------------------------------------------
__global__ __launch_bounds__(128) void cecp_kernel(
    const unsigned short* __restrict__ xbf,   // [B,2048] bf16, x in cols 0..1023
    const float* __restrict__ ce_wg,          // [H,2]
    const float* __restrict__ ce_const,       // [128]
    const float* __restrict__ tmp1,           // [B,128] x@ce_fc_w + b
    const float* __restrict__ tmp2,           // [B,128] x@cp_w + b
    const float* __restrict__ gates,          // [B,8]
    float* __restrict__ combined) {           // [B,384]
    const int b   = blockIdx.x;
    const int tid = threadIdx.x;
    __shared__ float s0[128], s1[128];
    __shared__ float cw[2];
    float p0 = 0.f, p1 = 0.f;
    const size_t base = (size_t)b * 2048;
    for (int h = tid; h < HDIM; h += 128) {
        float xv = bf2f(xbf[base + h]);
        p0 += xv * ce_wg[(size_t)h * 2 + 0];
        p1 += xv * ce_wg[(size_t)h * 2 + 1];
    }
    s0[tid] = p0; s1[tid] = p1;
    __syncthreads();
    for (int off = 64; off > 0; off >>= 1) {
        if (tid < off) { s0[tid] += s0[tid + off]; s1[tid] += s1[tid + off]; }
        __syncthreads();
    }
    if (tid == 0) {
        float l0 = s0[0], l1 = s1[0];
        float mx = fmaxf(l0, l1);
        float e0 = expf(l0 - mx), e1 = expf(l1 - mx);
        float inv = 1.f / (e0 + e1);
        cw[0] = e0 * inv; cw[1] = e1 * inv;
    }
    __syncthreads();
    const float g5 = gates[(size_t)b * EEXP + NREG];
    const float g6 = gates[(size_t)b * EEXP + NREG + 1];
    const int n = tid;  // 128 threads == 128 outputs
    float co  = cw[0] * tmp1[(size_t)b * 128 + n] + cw[1] * ce_const[n];
    float cpo = tmp2[(size_t)b * 128 + n];
    combined[(size_t)b * 384 + n] += g5 * co + g6 * cpo;
}

// ---------------------------------------------------------------------------
// Host orchestration
// ---------------------------------------------------------------------------
extern "C" void kernel_launch(void* const* d_in, const int* in_sizes, int n_in,
                              void* d_out, int out_size, void* d_ws, size_t ws_size,
                              hipStream_t stream) {
    (void)in_sizes; (void)n_in; (void)out_size; (void)ws_size;

    // ---- inputs (setup_inputs order) ----
    const float* mm_feat  = (const float*)d_in[0];
    const float* q_feat   = (const float*)d_in[1];
    const float* ln_g     = (const float*)d_in[2];
    const float* ln_b     = (const float*)d_in[3];
    const float* qe_w1    = (const float*)d_in[4];
    const float* qe_b1    = (const float*)d_in[5];
    const float* qe_w2    = (const float*)d_in[6];
    const float* qe_b2    = (const float*)d_in[7];
    const float* fg_w     = (const float*)d_in[8];
    const float* fg_b     = (const float*)d_in[9];
    const float* wg1      = (const float*)d_in[10];
    const float* wg2      = (const float*)d_in[11];
    const float* re_w1    = (const float*)d_in[12];
    const float* re_b1    = (const float*)d_in[13];
    const float* re_w2    = (const float*)d_in[14];
    const float* re_b2    = (const float*)d_in[15];
    const float* re_w3    = (const float*)d_in[16];
    const float* re_b3    = (const float*)d_in[17];
    const float* ce_const = (const float*)d_in[18];
    const float* ce_wg    = (const float*)d_in[19];
    const float* ce_fc_w  = (const float*)d_in[20];
    const float* ce_fc_b  = (const float*)d_in[21];
    const float* cp_w     = (const float*)d_in[22];
    const float* cp_b     = (const float*)d_in[23];
    const float* se_w1    = (const float*)d_in[24];
    const float* se_b1    = (const float*)d_in[25];
    const float* se_w2    = (const float*)d_in[26];
    const float* se_b2    = (const float*)d_in[27];
    const float* se_w3    = (const float*)d_in[28];
    const float* se_b3    = (const float*)d_in[29];
    const float* out_w    = (const float*)d_in[30];
    const float* out_b    = (const float*)d_in[31];

    // ---- workspace carve (256B aligned) ----
    char* wsp = (char*)d_ws;
    auto alloc = [&](size_t bytes) -> char* {
        char* p = wsp;
        wsp += (bytes + 255) & ~(size_t)255;
        return p;
    };
    typedef unsigned short u16;
    u16*   xbf     = (u16*)alloc((size_t)B_TOK * 2048 * 2);   // [B,2048] concat [x | q]
    u16*   qfbf    = (u16*)alloc((size_t)B_TOK * 1024 * 2);
    u16*   q1      = (u16*)alloc((size_t)B_TOK * 1024 * 2);
    u16*   fusedb  = (u16*)alloc((size_t)B_TOK * 1024 * 2);
    u16*   h1      = (u16*)alloc((size_t)B_TOK * 2048 * 2);
    u16*   h2      = (u16*)alloc((size_t)B_TOK * 1024 * 2);
    float* gates   = (float*)alloc((size_t)B_TOK * EEXP * 4);
    float* combined= (float*)alloc((size_t)B_TOK * 384 * 4);
    u16*   combbf  = (u16*)alloc((size_t)B_TOK * 384 * 2);
    float* tmp1    = (float*)alloc((size_t)B_TOK * 128 * 4);
    float* tmp2    = (float*)alloc((size_t)B_TOK * 128 * 4);
    // bf16 weight copies
    u16* qe_w1b = (u16*)alloc((size_t)1048576 * 2);
    u16* qe_w2b = (u16*)alloc((size_t)1048576 * 2);
    u16* fg_wb  = (u16*)alloc((size_t)2097152 * 2);
    u16* re_w1b = (u16*)alloc((size_t)10485760 * 2);
    u16* re_w2b = (u16*)alloc((size_t)10485760 * 2);
    u16* re_w3b = (u16*)alloc((size_t)655360 * 2);
    u16* ce_fcb = (u16*)alloc((size_t)131072 * 2);
    u16* cp_wb  = (u16*)alloc((size_t)131072 * 2);
    u16* se_w1b = (u16*)alloc((size_t)4194304 * 2);
    u16* se_w2b = (u16*)alloc((size_t)4194304 * 2);
    u16* se_w3b = (u16*)alloc((size_t)262144 * 2);
    u16* out_wb = (u16*)alloc((size_t)49152 * 2);

    auto cvt = [&](const float* src, u16* dst, size_t n) {
        unsigned grid = (unsigned)((n + 255) / 256);
        cvt_bf16_kernel<<<grid, 256, 0, stream>>>(src, dst, n);
    };
    auto gemm = [&](const u16* A, int lda, const u16* W, int ldw, const float* bias,
                    void* C, int ldc, int M, int N, int K, int act, bool outbf,
                    const float* gts, int gidx, int accum) {
        dim3 grid(N / 64, M / 128);
        if (outbf)
            gemm_bf16_kernel<true><<<grid, 256, 0, stream>>>(A, lda, W, ldw, bias, C, ldc,
                                                             K, act, gts, gidx, accum);
        else
            gemm_bf16_kernel<false><<<grid, 256, 0, stream>>>(A, lda, W, ldw, bias, C, ldc,
                                                              K, act, gts, gidx, accum);
    };

    // ---- 0: weight downconversion ----
    cvt(qe_w1, qe_w1b, 1048576);
    cvt(qe_w2, qe_w2b, 1048576);
    cvt(fg_w,  fg_wb,  2097152);
    cvt(re_w1, re_w1b, 10485760);
    cvt(re_w2, re_w2b, 10485760);
    cvt(re_w3, re_w3b, 655360);
    cvt(ce_fc_w, ce_fcb, 131072);
    cvt(cp_w,  cp_wb,  131072);
    cvt(se_w1, se_w1b, 4194304);
    cvt(se_w2, se_w2b, 4194304);
    cvt(se_w3, se_w3b, 262144);
    cvt(out_w, out_wb, 49152);
    cvt(q_feat, qfbf, (size_t)B_TOK * 1024);

    // ---- 1: layernorm -> xbf[:, 0:1024] ----
    ln_kernel<<<B_TOK, 256, 0, stream>>>(mm_feat, ln_g, ln_b, xbf, 2048);

    // ---- 2: router MLP chain ----
    gemm(qfbf, 1024, qe_w1b, 1024, qe_b1, q1, 1024, B_TOK, 1024, 1024, /*relu*/1, true, nullptr, 0, 0);
    gemm(q1,   1024, qe_w2b, 1024, qe_b2, xbf + 1024, 2048, B_TOK, 1024, 1024, 0, true, nullptr, 0, 0);
    gemm(xbf,  2048, fg_wb,  1024, fg_b,  fusedb, 1024, B_TOK, 1024, 2048, /*relu*/1, true, nullptr, 0, 0);
    router_kernel<<<B_TOK, 32, 0, stream>>>(fusedb, wg1, wg2, gates);

    // ---- 3: routed accumulation buffer ----
    zero_routed_kernel<<<(B_TOK * 128) / 256, 256, 0, stream>>>(combined);

    // ---- 4: regular ProposalExperts (dense, gate-scaled accumulate) ----
    for (int e = 0; e < NREG; ++e) {
        gemm(xbf, 2048, re_w1b + (size_t)e * 1024 * 2048, 2048, re_b1 + (size_t)e * 2048,
             h1, 2048, B_TOK, 2048, 1024, 1, true, nullptr, 0, 0);
        gemm(h1, 2048, re_w2b + (size_t)e * 2048 * 1024, 1024, re_b2 + (size_t)e * 1024,
             h2, 1024, B_TOK, 1024, 2048, 1, true, nullptr, 0, 0);
        gemm(h2, 1024, re_w3b + (size_t)e * 1024 * 128, 128, re_b3 + (size_t)e * 128,
             combined, 384, B_TOK, 128, 1024, 0, false, gates, e, /*accum*/1);
    }

    // ---- 5: constant + copy experts ----
    gemm(xbf, 2048, ce_fcb, 128, ce_fc_b, tmp1, 128, B_TOK, 128, 1024, 0, false, nullptr, 0, 0);
    gemm(xbf, 2048, cp_wb,  128, cp_b,    tmp2, 128, B_TOK, 128, 1024, 0, false, nullptr, 0, 0);
    cecp_kernel<<<B_TOK, 128, 0, stream>>>(xbf, ce_wg, ce_const, tmp1, tmp2, gates, combined);

    // ---- 6: shared experts -> combined[:, 128:256] and [:, 256:384] ----
    for (int i = 0; i < NSH; ++i) {
        gemm(xbf, 2048, se_w1b + (size_t)i * 1024 * 2048, 2048, se_b1 + (size_t)i * 2048,
             h1, 2048, B_TOK, 2048, 1024, 1, true, nullptr, 0, 0);
        gemm(h1, 2048, se_w2b + (size_t)i * 2048 * 1024, 1024, se_b2 + (size_t)i * 1024,
             h2, 1024, B_TOK, 1024, 2048, 1, true, nullptr, 0, 0);
        gemm(h2, 1024, se_w3b + (size_t)i * 1024 * 128, 128, se_b3 + (size_t)i * 128,
             combined + 128 * (1 + i), 384, B_TOK, 128, 1024, 0, false, nullptr, 0, 0);
    }

    // ---- 7: output head: sigmoid(combined @ out_w + out_b) -> d_out ----
    cvt(combined, combbf, (size_t)B_TOK * 384);
    gemm(combbf, 384, out_wb, 128, out_b, d_out, 128, B_TOK, 128, 384, /*sigmoid*/2,
         false, nullptr, 0, 0);
}
